// GawKAN_29996051595934
// MI455X (gfx1250) — matrix operand order repaired and checked by reading, not compiled, not run on hardware
//
#include <hip/hip_runtime.h>
#include <hip/hip_bf16.h>

// ---------------------------------------------------------------------------
// Types for CDNA5 WMMA (wave32): V_WMMA_F32_16X16X32_BF16
// ---------------------------------------------------------------------------
typedef __attribute__((ext_vector_type(16))) __bf16 bf16x16;
typedef __attribute__((ext_vector_type(4)))  __bf16 bf16x4;
typedef __attribute__((ext_vector_type(2)))  __bf16 bf16x2;
typedef __attribute__((ext_vector_type(8)))  float  f32x8;

#define TILE_M 128
#define TILE_N 64
#define TILE_K 32
#define ASTR   40   // padded LDS row stride (elements): 80B -> conflict-free frags
#define BSTR   40

// ---------------------------------------------------------------------------
// Generic fused GEMM:  C = op_A(A) @ B  (+bias) (+resid)
//   AFFINE: op_A(A)[m,k] = A[m,k] * scaleA[k] + biasA[k]
//   batched via blockIdx.z with element strides sA/sB/sC (adj GEMM: sA = 0).
// A: M x K (row major, lda), B: K x N (row major, ldb), C: M x N (ldc).
// All dims are multiples of the tile (true for every call below).
// Block: 256 threads = 8 waves; wave w computes rows [w*16, w*16+16) x all 64
// columns -> 4 f32 16x16 accumulators sharing one A fragment.
// Software pipeline: store tile k to LDS -> barrier -> issue global loads for
// tile k+1 into registers -> WMMA on tile k -> barrier.  Global latency hides
// under the WMMA+DS phase; every fragment is 2 contiguous ds_load_b128.
// ---------------------------------------------------------------------------
template <bool AFFINE>
__global__ __launch_bounds__(256)
void gemm_bf16_wmma(const float* __restrict__ A, int lda, long sA,
                    const float* __restrict__ B, int ldb, long sB,
                    float*       __restrict__ C, int ldc, long sC,
                    const float* __restrict__ bias,
                    const float* __restrict__ scaleA,
                    const float* __restrict__ biasA,
                    const float* __restrict__ resid, int ldres,
                    int M, int N, int K)
{
    __shared__ __bf16 As[TILE_M * ASTR];   // [m][k] (padded)      10.0 KB
    __shared__ __bf16 Bs[TILE_N * BSTR];   // [n][k] transposed     5.0 KB

    A += (long)blockIdx.z * sA;
    B += (long)blockIdx.z * sB;
    C += (long)blockIdx.z * sC;

    const int t    = threadIdx.x;
    const int wave = t >> 5;          // 0..7 : row strip
    const int lane = t & 31;
    const int m0   = blockIdx.y * TILE_M;
    const int n0   = blockIdx.x * TILE_N;

    // staging registers (double-buffer against LDS)
    float4 va[4], vb[2], sc[4], bi[4];
    const int kp = t >> 4;            // 0..15 : B k-pair
    const int n4 = (t & 15) << 2;     // 0..60 : B column group

    auto load_stage = [&](int kk) {
#pragma unroll
        for (int e = 0; e < 4; ++e) {
            int idx = e * 256 + t;               // float4 index (128 rows x 8)
            int r   = idx >> 3;
            int k4  = (idx & 7) << 2;
            va[e] = *reinterpret_cast<const float4*>(
                &A[(long)(m0 + r) * lda + (kk + k4)]);
            if constexpr (AFFINE) {
                sc[e] = *reinterpret_cast<const float4*>(&scaleA[kk + k4]);
                bi[e] = *reinterpret_cast<const float4*>(&biasA[kk + k4]);
            }
        }
        vb[0] = *reinterpret_cast<const float4*>(
            &B[(long)(kk + 2 * kp)     * ldb + (n0 + n4)]);
        vb[1] = *reinterpret_cast<const float4*>(
            &B[(long)(kk + 2 * kp + 1) * ldb + (n0 + n4)]);
    };

    auto store_stage = [&]() {
#pragma unroll
        for (int e = 0; e < 4; ++e) {
            int idx = e * 256 + t;
            int r   = idx >> 3;
            int k4  = (idx & 7) << 2;
            float4 v = va[e];
            if constexpr (AFFINE) {
                v.x = v.x * sc[e].x + bi[e].x;  v.y = v.y * sc[e].y + bi[e].y;
                v.z = v.z * sc[e].z + bi[e].z;  v.w = v.w * sc[e].w + bi[e].w;
            }
            bf16x4 p;
            p[0] = (__bf16)v.x; p[1] = (__bf16)v.y;
            p[2] = (__bf16)v.z; p[3] = (__bf16)v.w;
            *reinterpret_cast<bf16x4*>(&As[r * ASTR + k4]) = p;
        }
        float c0[4] = {vb[0].x, vb[0].y, vb[0].z, vb[0].w};
        float c1[4] = {vb[1].x, vb[1].y, vb[1].z, vb[1].w};
#pragma unroll
        for (int i = 0; i < 4; ++i) {
            bf16x2 p;
            p[0] = (__bf16)c0[i];                // (k even, n)
            p[1] = (__bf16)c1[i];                // (k odd , n)
            *reinterpret_cast<bf16x2*>(&Bs[(n4 + i) * BSTR + 2 * kp]) = p;
        }
    };

    f32x8 acc[4];
#pragma unroll
    for (int j = 0; j < 4; ++j) acc[j] = (f32x8){};

    load_stage(0);                               // pipeline prologue

    for (int k0 = 0; k0 < K; k0 += TILE_K) {
        store_stage();
        __syncthreads();

        // kick off next tile's global loads; they overlap the WMMA phase
        if (k0 + TILE_K < K) load_stage(k0 + TILE_K);

        // ---- A fragment per documented 16-bit A layout:
        // lanes 0-15 -> K {0..7, 16..23}; lanes 16-31 -> K {8..15, 24..31}
        const int am = wave * 16 + (lane & 15);
        const int ah = (lane >> 4) << 3;         // 0 or 8
        bf16x16 a;
#pragma unroll
        for (int i = 0; i < 8; ++i) a[i]     = As[am * ASTR + ah + i];
#pragma unroll
        for (int i = 0; i < 8; ++i) a[8 + i] = As[am * ASTR + 16 + ah + i];

        // ---- all B fragments up-front so DS returns overlap WMMA issue
        const int kb = (lane >> 4) << 3;         // 0 or 8
        const int nl = lane & 15;
        bf16x16 bv[4];
#pragma unroll
        for (int j = 0; j < 4; ++j) {
            const int nrow = (j * 16 + nl) * BSTR + kb;
#pragma unroll
            for (int i = 0; i < 8; ++i) bv[j][i] = Bs[nrow + i];
#pragma unroll
            for (int i = 0; i < 8; ++i) bv[j][8 + i] = Bs[nrow + 16 + i];
        }
#pragma unroll
        for (int j = 0; j < 4; ++j)
            acc[j] = __builtin_amdgcn_wmma_f32_16x16x32_bf16(
                false, a, false, bv[j], (short)0, acc[j], false, false);
        __syncthreads();
    }

    // ---- epilogue: C/D layout: lanes 0-15 M=r, lanes 16-31 M=r+8; N = lane&15
    const int lm = (lane >> 4) << 3;
#pragma unroll
    for (int j = 0; j < 4; ++j) {
        const int col  = n0 + j * 16 + (lane & 15);
        const float bc = bias ? bias[col] : 0.f;
#pragma unroll
        for (int r = 0; r < 8; ++r) {
            int row  = m0 + wave * 16 + lm + r;
            float v  = acc[j][r] + bc;
            if (resid) v += resid[(long)row * ldres + col];
            C[(long)row * ldc + col] = v;
        }
    }
}

// ---------------------------------------------------------------------------
// Wavelet coefficient: coeff[f] = Wnl[f] . P[argmax_l(logits[l]+gumb[f,l])]
// Haar/db1 'zero' patterns for MAX_LEVEL=3, FULL_LEN=8 (hard-coded).
// ---------------------------------------------------------------------------
__global__ void coeff_kernel(const float* __restrict__ logits,
                             const float* __restrict__ gumb,
                             const float* __restrict__ Wnl,
                             float* __restrict__ coeff)
{
    const float P[3][8] = {
        {0.70710678f, 0.70710678f, 0.f, 0.f,  0.f, 0.f,  0.f, 0.f},
        {0.5f,        0.70710678f, 0.f, 0.5f, 0.f, 0.f,  0.f, 0.f},
        {0.35355339f, 0.70710678f, 0.f, 0.f,  0.f, 0.5f, 0.f, 0.35355339f}
    };
    int f = threadIdx.x;                       // 256 features
    float best = logits[0] + gumb[f * 3 + 0];
    int   bj   = 0;
#pragma unroll
    for (int j = 1; j < 3; ++j) {
        float v = logits[j] + gumb[f * 3 + j];
        if (v > best) { best = v; bj = j; }
    }
    float acc = 0.f;
#pragma unroll
    for (int i = 0; i < 8; ++i) acc += Wnl[f * 8 + i] * P[bj][i];
    coeff[f] = acc;
}

// ---------------------------------------------------------------------------
// _mix support: sum((a-b)^2) -> partials -> sigmoid(mean) -> blend
// (float4-vectorized streaming; all counts divide 4, buffers 16B aligned)
// ---------------------------------------------------------------------------
__global__ void reduce_diffsq(const float4* __restrict__ a,
                              const float4* __restrict__ b,
                              long count4, float* __restrict__ part)
{
    __shared__ float s[256];
    float acc = 0.f;
    long stride = (long)gridDim.x * blockDim.x;
    for (long i = (long)blockIdx.x * blockDim.x + threadIdx.x; i < count4; i += stride) {
        float4 va = a[i], vb = b[i];
        float dx = va.x - vb.x, dy = va.y - vb.y;
        float dz = va.z - vb.z, dw = va.w - vb.w;
        acc += dx * dx + dy * dy + dz * dz + dw * dw;
    }
    s[threadIdx.x] = acc;
    __syncthreads();
    for (int st = 128; st > 0; st >>= 1) {
        if ((int)threadIdx.x < st) s[threadIdx.x] += s[threadIdx.x + st];
        __syncthreads();
    }
    if (threadIdx.x == 0) part[blockIdx.x] = s[0];
}

__global__ void finalize_th(const float* __restrict__ part, int n,
                            float invcnt, float* __restrict__ th_out)
{
    __shared__ float s[256];
    float acc = 0.f;
    for (int i = threadIdx.x; i < n; i += 256) acc += part[i];
    s[threadIdx.x] = acc;
    __syncthreads();
    for (int st = 128; st > 0; st >>= 1) {
        if ((int)threadIdx.x < st) s[threadIdx.x] += s[threadIdx.x + st];
        __syncthreads();
    }
    if (threadIdx.x == 0) {
        float mean = s[0] * invcnt;
        th_out[0] = 1.f / (1.f + __expf(-mean));
    }
}

__global__ void mix_blend(float4* __restrict__ h, const float4* __restrict__ r,
                          long count4, const float* __restrict__ thp)
{
    float th = thp[0];
    float om = 1.f - th;
    long stride = (long)gridDim.x * blockDim.x;
    for (long i = (long)blockIdx.x * blockDim.x + threadIdx.x; i < count4; i += stride) {
        float4 vh = h[i], vr = r[i];
        vh.x = th * vh.x + om * vr.x;  vh.y = th * vh.y + om * vr.y;
        vh.z = th * vh.z + om * vr.z;  vh.w = th * vh.w + om * vr.w;
        h[i] = vh;
    }
}

// ---------------------------------------------------------------------------
// Host-side orchestration
// ---------------------------------------------------------------------------
static inline void run_gemm(hipStream_t stream,
                            const float* A, int lda, long sA,
                            const float* B, int ldb, long sB,
                            float* C, int ldc, long sC,
                            const float* bias,
                            const float* scaleA, const float* biasA,
                            const float* resid, int ldres,
                            int M, int N, int K, int batch)
{
    dim3 grid(N / TILE_N, M / TILE_M, batch);
    if (scaleA)
        gemm_bf16_wmma<true><<<grid, 256, 0, stream>>>(
            A, lda, sA, B, ldb, sB, C, ldc, sC,
            bias, scaleA, biasA, resid, ldres, M, N, K);
    else
        gemm_bf16_wmma<false><<<grid, 256, 0, stream>>>(
            A, lda, sA, B, ldb, sB, C, ldc, sC,
            bias, scaleA, biasA, resid, ldres, M, N, K);
}

static inline void run_mix(hipStream_t stream, float* h, const float* r,
                           long count, float* part, float* th)
{
    reduce_diffsq<<<1024, 256, 0, stream>>>((const float4*)h, (const float4*)r,
                                            count / 4, part);
    finalize_th<<<1, 256, 0, stream>>>(part, 1024, 1.0f / (float)count, th);
    mix_blend<<<2048, 256, 0, stream>>>((float4*)h, (const float4*)r,
                                        count / 4, th);
}

extern "C" void kernel_launch(void* const* d_in, const int* in_sizes, int n_in,
                              void* d_out, int out_size, void* d_ws, size_t ws_size,
                              hipStream_t stream)
{
    (void)in_sizes; (void)n_in; (void)out_size; (void)ws_size;

    const float* x       = (const float*)d_in[0];
    const float* adj     = (const float*)d_in[1];
    const float* Wres    = (const float*)d_in[2];
    const float* bres    = (const float*)d_in[3];
    const float* Wg0     = (const float*)d_in[4];
    const float* bg0     = (const float*)d_in[5];
    const float* Wrg0    = (const float*)d_in[6];
    const float* brg0    = (const float*)d_in[7];
    const float* logits0 = (const float*)d_in[8];
    const float* gumb0   = (const float*)d_in[9];
    const float* Wnl0    = (const float*)d_in[10];
    const float* bnl0    = (const float*)d_in[11];
    const float* Wwo0    = (const float*)d_in[12];
    const float* bwo0    = (const float*)d_in[13];
    const float* Wrw0    = (const float*)d_in[14];
    const float* brw0    = (const float*)d_in[15];
    const float* Wg1     = (const float*)d_in[16];
    const float* bg1     = (const float*)d_in[17];
    const float* Wrg1    = (const float*)d_in[18];
    const float* brg1    = (const float*)d_in[19];
    const float* logits1 = (const float*)d_in[20];
    const float* gumb1   = (const float*)d_in[21];
    const float* Wnl1    = (const float*)d_in[22];
    const float* bnl1    = (const float*)d_in[23];
    const float* Wwo1    = (const float*)d_in[24];
    const float* bwo1    = (const float*)d_in[25];
    const float* Wrw1    = (const float*)d_in[26];
    const float* brw1    = (const float*)d_in[27];

    const long Bsz = 128, Nn = 512;
    const long M = Bsz * Nn;               // 65536 rows
    const int  Din = 64, H = 256, Dout = 64;

    float* ws    = (float*)d_ws;
    float* RES   = ws;                     // M*64   : x @ Wres + bres
    float* HH    = RES  + M * Din;         // M*256  : current hidden
    float* H2    = HH   + M * H;           // M*256  : secondary hidden
    float* TMP   = H2   + M * H;           // M*256  : graph-conv temp / res-proj
    float* coef0 = TMP  + M * H;           // 256
    float* coef1 = coef0 + 256;            // 256
    float* part  = coef1 + 256;            // 1024 partials
    float* scal  = part + 1024;            // 4 thresholds

    float* OUT = (float*)d_out;            // M*64

    // wavelet selection coefficients (hard Gumbel argmax forward value)
    coeff_kernel<<<1, 256, 0, stream>>>(logits0, gumb0, Wnl0, coef0);
    coeff_kernel<<<1, 256, 0, stream>>>(logits1, gumb1, Wnl1, coef1);

    // res = x @ Wres + bres                                (M x 64)
    run_gemm(stream, x, Din, 0, Wres, Dout, 0, RES, Dout, 0,
             bres, nullptr, nullptr, nullptr, 0, (int)M, Dout, Din, 1);

    // ---- layer 0 ----
    // TMP(ld64) = adj @ x  (batched over B, shared adj)
    run_gemm(stream, adj, (int)Nn, 0, x, Din, Nn * Din, TMP, Din, Nn * Din,
             nullptr, nullptr, nullptr, nullptr, 0, (int)Nn, Din, (int)Nn, (int)Bsz);
    // H = TMP @ Wg0 + bg0                                  (M x 256)
    run_gemm(stream, TMP, Din, 0, Wg0, H, 0, HH, H, 0,
             bg0, nullptr, nullptr, nullptr, 0, (int)M, H, Din, 1);
    // TMP = res @ Wrg0 + brg0 ; mix
    run_gemm(stream, RES, Dout, 0, Wrg0, H, 0, TMP, H, 0,
             brg0, nullptr, nullptr, nullptr, 0, (int)M, H, Dout, 1);
    run_mix(stream, HH, TMP, M * H, part, scal + 0);
    // H2 = (H*coef0 + bnl0) @ Wwo0 + bwo0 + H              (wavelet + residual)
    run_gemm(stream, HH, H, 0, Wwo0, H, 0, H2, H, 0,
             bwo0, coef0, bnl0, HH, H, (int)M, H, H, 1);
    // TMP = res @ Wrw0 + brw0 ; mix
    run_gemm(stream, RES, Dout, 0, Wrw0, H, 0, TMP, H, 0,
             brw0, nullptr, nullptr, nullptr, 0, (int)M, H, Dout, 1);
    run_mix(stream, H2, TMP, M * H, part, scal + 1);

    // ---- layer 1 ----
    // TMP(ld256) = adj @ H2  (batched)
    run_gemm(stream, adj, (int)Nn, 0, H2, H, Nn * H, TMP, H, Nn * H,
             nullptr, nullptr, nullptr, nullptr, 0, (int)Nn, H, (int)Nn, (int)Bsz);
    // H = TMP @ Wg1 + bg1 + H2                             (residual add)
    run_gemm(stream, TMP, H, 0, Wg1, H, 0, HH, H, 0,
             bg1, nullptr, nullptr, H2, H, (int)M, H, H, 1);
    // TMP = res @ Wrg1 + brg1 ; mix
    run_gemm(stream, RES, Dout, 0, Wrg1, H, 0, TMP, H, 0,
             brg1, nullptr, nullptr, nullptr, 0, (int)M, H, Dout, 1);
    run_mix(stream, HH, TMP, M * H, part, scal + 2);
    // OUT = (H*coef1 + bnl1) @ Wwo1 + bwo1                 (256 -> 64)
    run_gemm(stream, HH, H, 0, Wwo1, Dout, 0, OUT, Dout, 0,
             bwo1, coef1, bnl1, nullptr, 0, (int)M, Dout, H, 1);
    // TMP(ld64) = res @ Wrw1 + brw1 ; mix into OUT
    run_gemm(stream, RES, Dout, 0, Wrw1, Dout, 0, TMP, Dout, 0,
             brw1, nullptr, nullptr, nullptr, 0, (int)M, Dout, Dout, 1);
    run_mix(stream, OUT, TMP, M * Dout, part, scal + 3);
}
